// BigBirdModel_3040836845890
// MI455X (gfx1250) — matrix operand order repaired and checked by reading, not compiled, not run on hardware
//
#include <hip/hip_runtime.h>
#include <math.h>

// Problem constants (B=1)
#define NSEQ 4096
#define DM   768
#define HH   12
#define DHD  64
#define FFD  3072
#define NLAY 4
#define MBLK 64   // NSEQ / BS
#define BS   64
#define RR   3

typedef __bf16 bf16;
typedef bf16  v16bf __attribute__((ext_vector_type(16)));
typedef bf16  v8bf  __attribute__((ext_vector_type(8)));
typedef float v8f   __attribute__((ext_vector_type(8)));
typedef unsigned int v4u __attribute__((ext_vector_type(4)));
typedef int v8i __attribute__((ext_vector_type(8)));
typedef int v4i __attribute__((ext_vector_type(4)));

#if __has_builtin(__builtin_amdgcn_tensor_load_to_lds) && \
    __has_builtin(__builtin_amdgcn_s_wait_tensorcnt)
#define HAVE_TDM 1
#else
#define HAVE_TDM 0
#endif

// ---------------------------------------------------------------------------
// WMMA helpers (CDNA5 wave32, 16x16x32 bf16 -> f32)
// ---------------------------------------------------------------------------
__device__ inline v8f wmma_bf16(v16bf a, v16bf b, v8f c) {
  return __builtin_amdgcn_wmma_f32_16x16x32_bf16(false, a, false, b, (short)0, c,
                                                 false, false);
}

// A fragment: 16x32 bf16 tile, row-major in LDS with row stride `lda`.
// ISA layout: lanes 0-15 -> M=lane, K-halves 0..7/16..23 ; lanes 16-31 -> 8..15/24..31
__device__ inline v16bf frag_a_load(const bf16* base, int lda) {
  const int lane = threadIdx.x & 31;
  const int row  = lane & 15;
  const int kh   = (lane >> 4) & 1;
  const bf16* p  = base + row * lda;
  v8bf lo = *(const v8bf*)(p + kh * 8);
  v8bf hi = *(const v8bf*)(p + 16 + kh * 8);
  return __builtin_shufflevector(lo, hi, 0, 1, 2, 3, 4, 5, 6, 7,
                                 8, 9, 10, 11, 12, 13, 14, 15);
}

// B fragment: 32x16 bf16 tile stored TRANSPOSED in LDS as [n][k] with row
// stride `ldn` so the 16 contraction values per lane are contiguous.
// ISA layout: lanes 0-15 -> K 0..7/16..23, lanes 16-31 -> K 8..15/24..31, col = lane%16.
__device__ inline v16bf frag_b_load(const bf16* base, int ldn) {
  const int lane = threadIdx.x & 31;
  const int col  = lane & 15;
  const int kb   = ((lane >> 4) & 1) * 8;
  const bf16* p  = base + col * ldn + kb;
  v8bf lo = *(const v8bf*)(p);
  v8bf hi = *(const v8bf*)(p + 16);
  return __builtin_shufflevector(lo, hi, 0, 1, 2, 3, 4, 5, 6, 7,
                                 8, 9, 10, 11, 12, 13, 14, 15);
}

// ---------------------------------------------------------------------------
// f32 -> bf16 bulk convert (weights / activations)
// ---------------------------------------------------------------------------
__global__ void cvt_bf16_kernel(const float* __restrict__ in, bf16* __restrict__ out,
                                size_t n) {
  size_t i = (size_t)blockIdx.x * blockDim.x + threadIdx.x;
  const size_t stride = (size_t)gridDim.x * blockDim.x;
  for (; i < n; i += stride) out[i] = (bf16)in[i];
}

// ---------------------------------------------------------------------------
// GEMM: C[M,N] = A[M,K] (bf16) * B[K,N] (bf16) + bias, optional GELU epilogue.
// Block tile 128x64, 8 wave32s in a 4x2 grid, each wave 32x32 (2x2 WMMA tiles).
// A tile is staged into LDS by the Tensor Data Mover (TENSORcnt-tracked);
// B tile is staged manually (transposed to [n][k] for contiguous fragments).
// ---------------------------------------------------------------------------
__global__ __launch_bounds__(256)
void gemm_bf16_kernel(const bf16* __restrict__ A, const bf16* __restrict__ Bw,
                      const float* __restrict__ bias,
                      float* Cf, bf16* Cb,
                      int Mdim, int Ndim, int Kdim, int epi) {
  // Single LDS block so As sits at LDS offset 0 (needed for D#.lds_addr).
  __shared__ __align__(16) bf16 smem[128 * 32 + 64 * 32];
  bf16* As = smem;             // [128][32], LDS byte offset 0
  bf16* Bs = smem + 128 * 32;  // [64][32] transposed ([n][k])

  const int tid  = threadIdx.x;
  const int lane = tid & 31;
  const int wid  = tid >> 5;
  const int wr   = wid & 3;   // wave row (32 rows each)
  const int wc   = wid >> 2;  // wave col (32 cols each)
  const int m0   = blockIdx.y * 128;
  const int n0   = blockIdx.x * 64;

  v8f acc[2][2];
#pragma unroll
  for (int i = 0; i < 2; ++i)
#pragma unroll
    for (int j = 0; j < 2; ++j)
#pragma unroll
      for (int e = 0; e < 8; ++e) acc[i][j][e] = 0.f;

  const int bk = tid >> 3, bn = (tid & 7) * 8;   // B staging: 8 elems/thread
#if !HAVE_TDM
  const int ar = tid >> 1, ac = (tid & 1) * 16;  // A staging fallback
#endif

  for (int k0 = 0; k0 < Kdim; k0 += 32) {
    // ---- B tile: manual transpose staging --------------------------------
    const bf16* bg = Bw + (size_t)(k0 + bk) * Ndim + n0 + bn;
#pragma unroll
    for (int j = 0; j < 8; ++j) Bs[(bn + j) * 32 + bk] = bg[j];
    if (k0 + 32 < Kdim)
      __builtin_prefetch(Bw + (size_t)(k0 + 32 + bk) * Ndim + n0 + bn, 0, 1);

    // ---- A tile: Tensor Data Mover (wave 0 issues, TENSORcnt-tracked) ----
#if HAVE_TDM
    if (wid == 0) {
      const unsigned long long ga =
          (unsigned long long)(unsigned long long)(const void*)(A + (size_t)m0 * Kdim + k0);
      const unsigned dim0 = (unsigned)Kdim;          // tensor x-extent (elems)
      const unsigned dim1 = (unsigned)Mdim;          // tensor y-extent (rows)
      const unsigned long long st0 = (unsigned long long)(unsigned)Kdim;  // row stride

      v4u g0;
      g0[0] = 1u;                                           // count=1 (valid D#)
      g0[1] = 0u;                                           // lds_addr = 0 (As)
      g0[2] = (unsigned)(ga & 0xFFFFFFFFull);               // global_addr[31:0]
      g0[3] = (unsigned)((ga >> 32) & 0x1FFFFFFull) | (2u << 30);  // [56:32] | type=2

      v8i g1;
      g1[0] = (int)(1u << 16);                              // data_size=1 (2B), mask=0
      g1[1] = (int)((dim0 & 0xFFFFu) << 16);                // tensor_dim0[15:0]
      g1[2] = (int)((dim0 >> 16) | ((dim1 & 0xFFFFu) << 16));  // dim0[31:16]|dim1[15:0]
      g1[3] = (int)((dim1 >> 16) | (32u << 16));            // dim1[31:16]|tile_dim0=32
      g1[4] = (int)128u;                                    // tile_dim1=128, tile_dim2=0
      g1[5] = (int)(st0 & 0xFFFFFFFFull);                   // dim0_stride[31:0]
      g1[6] = (int)((st0 >> 32) & 0xFFFFull);               // stride[47:32]|dim1_stride=0
      g1[7] = 0;

      v4i g2; g2[0] = 0; g2[1] = 0; g2[2] = 0; g2[3] = 0;   // 2D: dims 2/3 unused
      v4i g3; g3[0] = 0; g3[1] = 0; g3[2] = 0; g3[3] = 0;
      v8i g4;                                               // extra group (clang-23
      g4[0] = 0; g4[1] = 0; g4[2] = 0; g4[3] = 0;           // 6-arg form), zero-filled
      g4[4] = 0; g4[5] = 0; g4[6] = 0; g4[7] = 0;

      __builtin_amdgcn_tensor_load_to_lds(g0, g1, g2, g3, g4, 0);
      __builtin_amdgcn_s_wait_tensorcnt(0);
    }
#else
    const bf16* ag = A + (size_t)(m0 + ar) * Kdim + k0 + ac;
    *(v8bf*)(As + ar * 32 + ac)     = *(const v8bf*)(ag);
    *(v8bf*)(As + ar * 32 + ac + 8) = *(const v8bf*)(ag + 8);
    if (k0 + 32 < Kdim)
      __builtin_prefetch(A + (size_t)(m0 + ar) * Kdim + k0 + 32 + ac, 0, 1);
#endif
    __syncthreads();

    v16bf af[2], bfm[2];
#pragma unroll
    for (int i = 0; i < 2; ++i) af[i] = frag_a_load(As + (wr * 32 + i * 16) * 32, 32);
#pragma unroll
    for (int j = 0; j < 2; ++j) bfm[j] = frag_b_load(Bs + (wc * 32 + j * 16) * 32, 32);
#pragma unroll
    for (int i = 0; i < 2; ++i)
#pragma unroll
      for (int j = 0; j < 2; ++j) acc[i][j] = wmma_bf16(af[i], bfm[j], acc[i][j]);
    __syncthreads();
  }

  // Epilogue: C layout -> VGPR e holds row (e + 8*(lane>=16)), col = lane%16
  const int rb = (lane < 16) ? 0 : 8;
  const int cc = lane & 15;
#pragma unroll
  for (int i = 0; i < 2; ++i)
#pragma unroll
    for (int j = 0; j < 2; ++j) {
      const int col = n0 + wc * 32 + j * 16 + cc;
      const float bi = bias ? bias[col] : 0.f;
#pragma unroll
      for (int e = 0; e < 8; ++e) {
        const int row = m0 + wr * 32 + i * 16 + rb + e;
        float v = acc[i][j][e] + bi;
        if (epi == 1) v = 0.5f * v * (1.0f + erff(v * 0.70710678118f));  // exact GELU
        const size_t off = (size_t)row * Ndim + col;
        if (Cf) Cf[off] = v;
        if (Cb) Cb[off] = (bf16)v;
      }
    }
}

// ---------------------------------------------------------------------------
// BigBird block-sparse attention. Grid (H, MBLK), 128 threads (4 wave32s).
// Each wave owns a 16-row query strip; streaming online-softmax over the
// key-block list. QK^T and P*V via WMMA bf16.
// q/k/v: bf16 [NSEQ][DM] head-major columns; out: bf16 [NSEQ][DM].
// ---------------------------------------------------------------------------
__global__ __launch_bounds__(128)
void bigbird_attn_kernel(const bf16* __restrict__ q, const bf16* __restrict__ k,
                         const bf16* __restrict__ v, const float* __restrict__ mask,
                         const int* __restrict__ rand_attn, bf16* __restrict__ out) {
  const int h    = blockIdx.x;
  const int qb   = blockIdx.y;
  const int tid  = threadIdx.x;
  const int lane = tid & 31;
  const int wave = tid >> 5;

  __shared__ __align__(16) bf16 Qs[BS * DHD];      // [qrow][dh] (pre-scaled)
  __shared__ __align__(16) bf16 Ks[BS * DHD];      // [key][dh]
  __shared__ __align__(16) bf16 Vt[DHD * BS];      // [dh][key] (transposed)
  __shared__ __align__(16) bf16 Ps[4 * 16 * BS];   // per-wave P strips

  // Key-block list (BigBird: global + window + random; dense first/last rows)
  int lst[8];
  int nkb;
  const bool dense = (qb == 0 || qb == MBLK - 1);
  if (dense) {
    nkb = MBLK;
  } else {
    const int* ra = rand_attn + ((size_t)h * (MBLK - 2) + (qb - 1)) * RR;
    if (qb == 1) {
      lst[0] = 0; lst[1] = 1; lst[2] = 2; lst[3] = MBLK - 1;
      lst[4] = ra[0]; lst[5] = ra[1]; lst[6] = ra[2]; nkb = 7;
    } else if (qb == MBLK - 2) {
      lst[0] = 0; lst[1] = MBLK - 3; lst[2] = MBLK - 2; lst[3] = MBLK - 1;
      lst[4] = ra[0]; lst[5] = ra[1]; lst[6] = ra[2]; nkb = 7;
    } else {
      lst[0] = 0; lst[1] = qb - 1; lst[2] = qb; lst[3] = qb + 1;
      lst[4] = MBLK - 1; lst[5] = ra[0]; lst[6] = ra[1]; lst[7] = ra[2]; nkb = 8;
    }
  }

  // Stage Q (fold in 1/sqrt(DH) = 0.125)
  {
    const int r = tid >> 1, c0 = (tid & 1) * 32;
    const bf16* src = q + (size_t)(qb * BS + r) * DM + h * DHD + c0;
#pragma unroll
    for (int j = 0; j < 32; ++j) Qs[r * DHD + c0 + j] = (bf16)((float)src[j] * 0.125f);
  }
  __syncthreads();

  v16bf aq[2];
#pragma unroll
  for (int kc = 0; kc < 2; ++kc)
    aq[kc] = frag_a_load(Qs + (wave * 16) * DHD + kc * 32, DHD);

  float mi[8], li[8];
  v8f o[4];
#pragma unroll
  for (int e = 0; e < 8; ++e) { mi[e] = -1e30f; li[e] = 0.f; }
#pragma unroll
  for (int d = 0; d < 4; ++d)
#pragma unroll
    for (int e = 0; e < 8; ++e) o[d][e] = 0.f;

  const int rb = (lane < 16) ? 0 : 8;
  const int cc = lane & 15;

  for (int it = 0; it < nkb; ++it) {
    const int kb = dense ? it : lst[it];

    // Stage K ([key][dh]) and V transposed ([dh][key])
    {
      const int r = tid >> 1, c0 = (tid & 1) * 32;
      const bf16* ks = k + (size_t)(kb * BS + r) * DM + h * DHD + c0;
      const bf16* vs = v + (size_t)(kb * BS + r) * DM + h * DHD + c0;
#pragma unroll
      for (int j = 0; j < 32; ++j) {
        Ks[r * DHD + c0 + j] = ks[j];
        Vt[(c0 + j) * BS + r] = vs[j];
      }
    }
    __syncthreads();

    // S strip (16 q-rows x 64 keys) = Q * K^T
    v8f s[4];
#pragma unroll
    for (int nt = 0; nt < 4; ++nt) {
#pragma unroll
      for (int e = 0; e < 8; ++e) s[nt][e] = 0.f;
#pragma unroll
      for (int kc = 0; kc < 2; ++kc)
        s[nt] = wmma_bf16(aq[kc], frag_b_load(Ks + (nt * 16) * DHD + kc * 32, DHD),
                          s[nt]);
      const float pen = (1.0f - mask[kb * BS + nt * 16 + cc]) * -10000.0f;
#pragma unroll
      for (int e = 0; e < 8; ++e) s[nt][e] += pen;
    }

    // Online softmax: each lane holds 8 rows; row values live in 16 lanes
    float mnew[8], alpha[8];
#pragma unroll
    for (int e = 0; e < 8; ++e) {
      float vmx = fmaxf(fmaxf(s[0][e], s[1][e]), fmaxf(s[2][e], s[3][e]));
#pragma unroll
      for (int off = 1; off < 16; off <<= 1) vmx = fmaxf(vmx, __shfl_xor(vmx, off, 32));
      mnew[e]  = fmaxf(mi[e], vmx);
      alpha[e] = __expf(mi[e] - mnew[e]);
      mi[e]    = mnew[e];
      float rs = 0.f;
#pragma unroll
      for (int nt = 0; nt < 4; ++nt) {
        const float p = __expf(s[nt][e] - mnew[e]);
        s[nt][e] = p;
        rs += p;
      }
#pragma unroll
      for (int off = 1; off < 16; off <<= 1) rs += __shfl_xor(rs, off, 32);
      li[e] = li[e] * alpha[e] + rs;
    }
#pragma unroll
    for (int d = 0; d < 4; ++d)
#pragma unroll
      for (int e = 0; e < 8; ++e) o[d][e] *= alpha[e];

    // P: C-layout regs -> per-wave LDS strip (bf16) for A-fragment reload
#pragma unroll
    for (int nt = 0; nt < 4; ++nt)
#pragma unroll
      for (int e = 0; e < 8; ++e)
        Ps[(wave * 16 + rb + e) * BS + nt * 16 + cc] = (bf16)s[nt][e];
    __syncthreads();

    // O += P * V  (contraction over 64 keys, 2 WMMA K-steps)
#pragma unroll
    for (int kc = 0; kc < 2; ++kc) {
      v16bf ap = frag_a_load(Ps + (wave * 16) * BS + kc * 32, BS);
#pragma unroll
      for (int d = 0; d < 4; ++d)
        o[d] = wmma_bf16(ap, frag_b_load(Vt + (d * 16) * BS + kc * 32, BS), o[d]);
    }
    __syncthreads();
  }

  // Write context (already [token][h*DH+dh] = transposed-merged layout)
#pragma unroll
  for (int d = 0; d < 4; ++d)
#pragma unroll
    for (int e = 0; e < 8; ++e) {
      const int row = qb * BS + wave * 16 + rb + e;
      const int col = h * DHD + d * 16 + cc;
      out[(size_t)row * DM + col] = (bf16)(o[d][e] / (li[e] + 1e-20f));
    }
}

// ---------------------------------------------------------------------------
// Embedding gather + LayerNorm. One block per token row (768 cols, 256 thr).
// ---------------------------------------------------------------------------
__global__ __launch_bounds__(256)
void embed_ln_kernel(const int* __restrict__ ids, const float* __restrict__ emb_word,
                     const float* __restrict__ emb_pos, const float* __restrict__ g,
                     const float* __restrict__ b, float* __restrict__ xo,
                     bf16* __restrict__ xb) {
  const int row = blockIdx.x, tid = threadIdx.x;
  __shared__ float red[256];
  const int id = ids[row];
  float vals[3];
  float s = 0.f;
#pragma unroll
  for (int j = 0; j < 3; ++j) {
    const int c = tid + j * 256;
    const float v = emb_word[(size_t)id * DM + c] + emb_pos[(size_t)row * DM + c];
    vals[j] = v;
    s += v;
  }
  red[tid] = s;
  __syncthreads();
  for (int off = 128; off > 0; off >>= 1) {
    if (tid < off) red[tid] += red[tid + off];
    __syncthreads();
  }
  const float mu = red[0] * (1.0f / DM);
  __syncthreads();
  float vs = 0.f;
#pragma unroll
  for (int j = 0; j < 3; ++j) {
    const float d = vals[j] - mu;
    vs += d * d;
  }
  red[tid] = vs;
  __syncthreads();
  for (int off = 128; off > 0; off >>= 1) {
    if (tid < off) red[tid] += red[tid + off];
    __syncthreads();
  }
  const float rstd = rsqrtf(red[0] * (1.0f / DM) + 1e-5f);
#pragma unroll
  for (int j = 0; j < 3; ++j) {
    const int c = tid + j * 256;
    const float y = (vals[j] - mu) * rstd * g[c] + b[c];
    xo[(size_t)row * DM + c] = y;
    xb[(size_t)row * DM + c] = (bf16)y;
  }
}

// ---------------------------------------------------------------------------
// x = LayerNorm(x + t) ; also emits bf16 copy. One block per token row.
// ---------------------------------------------------------------------------
__global__ __launch_bounds__(256)
void add_ln_kernel(const float* __restrict__ x, const float* __restrict__ t,
                   const float* __restrict__ g, const float* __restrict__ b,
                   float* __restrict__ xo, bf16* __restrict__ xb) {
  const int row = blockIdx.x, tid = threadIdx.x;
  __shared__ float red[256];
  float vals[3];
  float s = 0.f;
#pragma unroll
  for (int j = 0; j < 3; ++j) {
    const int c = tid + j * 256;
    const float v = x[(size_t)row * DM + c] + t[(size_t)row * DM + c];
    vals[j] = v;
    s += v;
  }
  red[tid] = s;
  __syncthreads();
  for (int off = 128; off > 0; off >>= 1) {
    if (tid < off) red[tid] += red[tid + off];
    __syncthreads();
  }
  const float mu = red[0] * (1.0f / DM);
  __syncthreads();
  float vs = 0.f;
#pragma unroll
  for (int j = 0; j < 3; ++j) {
    const float d = vals[j] - mu;
    vs += d * d;
  }
  red[tid] = vs;
  __syncthreads();
  for (int off = 128; off > 0; off >>= 1) {
    if (tid < off) red[tid] += red[tid + off];
    __syncthreads();
  }
  const float rstd = rsqrtf(red[0] * (1.0f / DM) + 1e-5f);
#pragma unroll
  for (int j = 0; j < 3; ++j) {
    const int c = tid + j * 256;
    const float y = (vals[j] - mu) * rstd * g[c] + b[c];
    xo[(size_t)row * DM + c] = y;
    xb[(size_t)row * DM + c] = (bf16)y;
  }
}

// ---------------------------------------------------------------------------
// Host orchestration
// ---------------------------------------------------------------------------
extern "C" void kernel_launch(void* const* d_in, const int* in_sizes, int n_in,
                              void* d_out, int out_size, void* d_ws, size_t ws_size,
                              hipStream_t stream) {
  (void)in_sizes; (void)n_in; (void)out_size; (void)ws_size;

  const int*   input_ids = (const int*)d_in[0];
  const float* mask      = (const float*)d_in[1];
  const int*   rand_attn = (const int*)d_in[2];
  const float* emb_word  = (const float*)d_in[3];
  const float* emb_pos   = (const float*)d_in[4];
  const float* eg        = (const float*)d_in[5];
  const float* ebb       = (const float*)d_in[6];
  const float* Wq = (const float*)d_in[7];   const float* bq  = (const float*)d_in[8];
  const float* Wk = (const float*)d_in[9];   const float* bk  = (const float*)d_in[10];
  const float* Wv = (const float*)d_in[11];  const float* bv  = (const float*)d_in[12];
  const float* Wo = (const float*)d_in[13];  const float* bo  = (const float*)d_in[14];
  const float* g1 = (const float*)d_in[15];  const float* be1 = (const float*)d_in[16];
  const float* W1 = (const float*)d_in[17];  const float* b1  = (const float*)d_in[18];
  const float* W2 = (const float*)d_in[19];  const float* b2  = (const float*)d_in[20];
  const float* g2 = (const float*)d_in[21];  const float* be2 = (const float*)d_in[22];

  float* xf = (float*)d_out;  // master residual stream (f32) == final output

  // Workspace carve (256B aligned)
  char* wp = (char*)d_ws;
  auto carve = [&](size_t bytes) -> char* {
    char* p = wp;
    wp += (bytes + 255) & ~(size_t)255;
    return p;
  };
  const size_t DD = (size_t)DM * DM, DF = (size_t)DM * FFD;
  bf16* wq_bf = (bf16*)carve(NLAY * DD * sizeof(bf16));
  bf16* wk_bf = (bf16*)carve(NLAY * DD * sizeof(bf16));
  bf16* wv_bf = (bf16*)carve(NLAY * DD * sizeof(bf16));
  bf16* wo_bf = (bf16*)carve(NLAY * DD * sizeof(bf16));
  bf16* w1_bf = (bf16*)carve(NLAY * DF * sizeof(bf16));
  bf16* w2_bf = (bf16*)carve(NLAY * DF * sizeof(bf16));
  bf16* x_bf  = (bf16*)carve((size_t)NSEQ * DM * sizeof(bf16));
  bf16* q_bf  = (bf16*)carve((size_t)NSEQ * DM * sizeof(bf16));
  bf16* k_bf  = (bf16*)carve((size_t)NSEQ * DM * sizeof(bf16));
  bf16* v_bf  = (bf16*)carve((size_t)NSEQ * DM * sizeof(bf16));
  bf16* a_bf  = (bf16*)carve((size_t)NSEQ * DM * sizeof(bf16));
  bf16* f_bf  = (bf16*)carve((size_t)NSEQ * FFD * sizeof(bf16));
  float* t_f  = (float*)carve((size_t)NSEQ * DM * sizeof(float));

  // Convert weights to bf16 once per launch
  cvt_bf16_kernel<<<1024, 256, 0, stream>>>(Wq, wq_bf, NLAY * DD);
  cvt_bf16_kernel<<<1024, 256, 0, stream>>>(Wk, wk_bf, NLAY * DD);
  cvt_bf16_kernel<<<1024, 256, 0, stream>>>(Wv, wv_bf, NLAY * DD);
  cvt_bf16_kernel<<<1024, 256, 0, stream>>>(Wo, wo_bf, NLAY * DD);
  cvt_bf16_kernel<<<1024, 256, 0, stream>>>(W1, w1_bf, NLAY * DF);
  cvt_bf16_kernel<<<1024, 256, 0, stream>>>(W2, w2_bf, NLAY * DF);

  // Embedding + LN
  embed_ln_kernel<<<NSEQ, 256, 0, stream>>>(input_ids, emb_word, emb_pos, eg, ebb,
                                            xf, x_bf);

  const dim3 gD(DM / 64, NSEQ / 128);    // (12, 32)
  const dim3 gF(FFD / 64, NSEQ / 128);   // (48, 32)
  const dim3 gAtt(HH, MBLK);             // (12, 64)

  for (int l = 0; l < NLAY; ++l) {
    gemm_bf16_kernel<<<gD, 256, 0, stream>>>(x_bf, wq_bf + l * DD, bq + (size_t)l * DM,
                                             nullptr, q_bf, NSEQ, DM, DM, 0);
    gemm_bf16_kernel<<<gD, 256, 0, stream>>>(x_bf, wk_bf + l * DD, bk + (size_t)l * DM,
                                             nullptr, k_bf, NSEQ, DM, DM, 0);
    gemm_bf16_kernel<<<gD, 256, 0, stream>>>(x_bf, wv_bf + l * DD, bv + (size_t)l * DM,
                                             nullptr, v_bf, NSEQ, DM, DM, 0);
    bigbird_attn_kernel<<<gAtt, 128, 0, stream>>>(q_bf, k_bf, v_bf, mask, rand_attn,
                                                  a_bf);
    gemm_bf16_kernel<<<gD, 256, 0, stream>>>(a_bf, wo_bf + l * DD, bo + (size_t)l * DM,
                                             t_f, nullptr, NSEQ, DM, DM, 0);
    add_ln_kernel<<<NSEQ, 256, 0, stream>>>(xf, t_f, g1 + (size_t)l * DM,
                                            be1 + (size_t)l * DM, xf, x_bf);
    gemm_bf16_kernel<<<gF, 256, 0, stream>>>(x_bf, w1_bf + l * DF, b1 + (size_t)l * FFD,
                                             nullptr, f_bf, NSEQ, FFD, DM, 1);
    gemm_bf16_kernel<<<gD, 256, 0, stream>>>(f_bf, w2_bf + l * DF, b2 + (size_t)l * DM,
                                             t_f, nullptr, NSEQ, DM, FFD, 0);
    add_ln_kernel<<<NSEQ, 256, 0, stream>>>(xf, t_f, g2 + (size_t)l * DM,
                                            be2 + (size_t)l * DM, xf, x_bf);
  }
}
